// WindowAttention_7447473291269
// MI455X (gfx1250) — matrix operand order repaired and checked by reading, not compiled, not run on hardware
//
#include <hip/hip_runtime.h>
#include <hip/hip_bf16.h>

typedef __attribute__((ext_vector_type(16))) __bf16          v16bf;
typedef __attribute__((ext_vector_type(16))) unsigned short  v16us;
typedef __attribute__((ext_vector_type(8)))  float           v8f;

#define WS_N   49
#define NSQ    2401          // 49*49
#define CDIM   256
#define HEADS  8
#define HD     32

// LDS strides in ushort units (rows 16B-aligned)
#define XSTR   264           // x / O tile [64][256] (+8 pad)
#define QKVSTR 776           // qkv tile [64][768] (+8 pad)
#define PSTR   72            // per-head probs [64][64] (+8 pad)
#define VTSTR  72            // transposed V [256][64] (+8 pad)

#define X_ELEMS    (64 * XSTR)
#define QKV_ELEMS  (64 * QKVSTR)
#define P_ELEMS    (HEADS * 64 * PSTR)
#define VT_ELEMS   (CDIM * VTSTR)
#define SMEM_BYTES ((X_ELEMS + QKV_ELEMS + P_ELEMS + VT_ELEMS) * 2)  // 243,712 B

#define QKVW_ELEMS  (768 * 256)          // 196608
#define PROJW_ELEMS (256 * 256)          // 65536
#define BIASX_ELEMS (HEADS * NSQ)        // 19208

union Frag {
  v16us u;
  v16bf b;
  uint4 q[2];
};

// native bf16 convert (lowers to v_cvt/v_fma_mix*_bf16)
__device__ __forceinline__ unsigned short f2bf(float f) {
  __bf16 h = (__bf16)f;
  return __builtin_bit_cast(unsigned short, h);
}
__device__ __forceinline__ unsigned pk2bf(float lo, float hi) {
  return (unsigned)f2bf(lo) | ((unsigned)f2bf(hi) << 16);
}

// ------------- pre-pass 1: weights f32 -> bf16 into d_ws ----------------------
__global__ __launch_bounds__(256) void cvt_weights(
    const float* __restrict__ qkv_w, const float* __restrict__ proj_w,
    unsigned short* __restrict__ wq, unsigned short* __restrict__ wp) {
  int i4 = (blockIdx.x * 256 + threadIdx.x) * 4;
  if (i4 < QKVW_ELEMS) {
    float4 f = *(const float4*)(qkv_w + i4);
    *(uint2*)(wq + i4) = make_uint2(pk2bf(f.x, f.y), pk2bf(f.z, f.w));
  } else {
    int j = i4 - QKVW_ELEMS;
    if (j < PROJW_ELEMS) {
      float4 f = *(const float4*)(proj_w + j);
      *(uint2*)(wp + j) = make_uint2(pk2bf(f.x, f.y), pk2bf(f.z, f.w));
    }
  }
}

// ------------- pre-pass 2: expand relative-position bias to [8][49][49] -------
__global__ __launch_bounds__(256) void expand_bias(
    const float* __restrict__ btab, float* __restrict__ bx) {
  int i = blockIdx.x * 256 + threadIdx.x;
  if (i >= BIASX_ELEMS) return;
  int h = i / NSQ, nm = i % NSQ;
  int n = nm / WS_N, m = nm % WS_N;
  int r0 = n / 7 - m / 7 + 6;
  int r1 = n % 7 - m % 7 + 6;
  bx[i] = btab[(r0 * 13 + r1) * HEADS + h];
}

// ------------- fused window attention, one workgroup per window ---------------
__global__ __launch_bounds__(256) void swin_window_attn_fused(
    const float* __restrict__ x,              // [4096, 49, 256]
    const float* __restrict__ mask,           // [64, 49, 49]
    const unsigned short* __restrict__ wq,    // qkv_w bf16 [768, 256]
    const float* __restrict__ qkv_b,          // [768]
    const unsigned short* __restrict__ wp,    // proj_w bf16 [256, 256]
    const float* __restrict__ proj_b,         // [256]
    const float* __restrict__ biasx,          // expanded bias [8][49][49]
    float* __restrict__ out)                  // [4096, 49, 256]
{
  extern __shared__ unsigned short lds[];
  unsigned short* Xs   = lds;                               // also O after attn
  unsigned short* QKVs = lds + X_ELEMS;
  unsigned short* Ps   = lds + X_ELEMS + QKV_ELEMS;
  unsigned short* Vt   = lds + X_ELEMS + QKV_ELEMS + P_ELEMS;

  const int b    = blockIdx.x;
  const int tid  = threadIdx.x;
  // wave id is wave-uniform: force it scalar so all derived addressing is SALU
  const int wv   = __builtin_amdgcn_readfirstlane(tid) >> 5;
  const int lane = tid & 31;
  const int lm   = lane & 15;
  const int half = lane >> 4;

  // proj weights are consumed last — warm L2/L0 early
  __builtin_prefetch(wp + (size_t)tid * 256, 0, 1);

  // ---- Phase 0: stage x tile into LDS as bf16 (rows padded 49 -> 64)
  const float* xb = x + (size_t)b * WS_N * CDIM;
  for (int i4 = tid * 4; i4 < 64 * CDIM; i4 += 256 * 4) {
    int r = i4 >> 8, c = i4 & 255;
    uint2 pk;
    if (r < WS_N) {
      float4 f = *(const float4*)(xb + r * CDIM + c);
      pk = make_uint2(pk2bf(f.x, f.y), pk2bf(f.z, f.w));
    } else {
      pk = make_uint2(0u, 0u);
    }
    *(uint2*)&Xs[r * XSTR + c] = pk;
  }
  __syncthreads();

  // ---- Phase 1: QKV = x @ qkv_w^T + qkv_b   ([64,256] x [256,768])
  const float scale = 0.17677669529663687f;   // hd^-0.5
  for (int cti = 0; cti < 6; ++cti) {
    const int ct = wv * 6 + cti;              // scalar col tile 0..47
    const int cbase = ct * 16;
    Frag Bf[8];
    const unsigned short* wrow = wq + (size_t)(cbase + lm) * CDIM + half * 16;
#pragma unroll
    for (int kk = 0; kk < 8; ++kk) {
      const uint4* p = (const uint4*)(wrow + kk * 32);
      Bf[kk].q[0] = p[0];
      Bf[kk].q[1] = p[1];
    }

    const float bias = qkv_b[cbase + lm];
    const float sc = (ct < 16) ? scale : 1.0f;   // q columns absorb 1/sqrt(hd)

    for (int rt = 0; rt < 4; ++rt) {
      v8f acc;
#pragma unroll
      for (int r = 0; r < 8; ++r) acc[r] = bias;
      const int arow = rt * 16 + lm;
#pragma unroll
      for (int kk = 0; kk < 8; ++kk) {
        Frag Af;
        Af.q[0] = *(const uint4*)&Xs[arow * XSTR + kk * 32 + half * 16];
        Af.q[1] = *(const uint4*)&Xs[arow * XSTR + kk * 32 + half * 16 + 8];
        acc = __builtin_amdgcn_wmma_f32_16x16x32_bf16(
            false, Af.b, false, Bf[kk].b, (short)0, acc, false, false);
      }
      if (ct >= 32) {                         // V tiles: also store transposed
        const int vd0 = cbase - 512;
#pragma unroll
        for (int r = 0; r < 8; ++r) {
          int row = rt * 16 + r + half * 8;
          unsigned short hv = f2bf(acc[r]);
          QKVs[row * QKVSTR + cbase + lm] = hv;
          Vt[(vd0 + lm) * VTSTR + row] = hv;
        }
      } else {
#pragma unroll
        for (int r = 0; r < 8; ++r) {
          int row = rt * 16 + r + half * 8;
          QKVs[row * QKVSTR + cbase + lm] = f2bf(acc[r] * sc);
        }
      }
    }
  }
  __syncthreads();

  // ---- Phase 2: per-head attention (wave wv == head h, scalar)
  const int h = wv;
  const int qoff = h * HD, koff = 256 + h * HD;

  // S = Q @ K^T  (K-dim = 32 -> one WMMA per 16x16 tile)
  v8f S[4][4];
  for (int rt = 0; rt < 4; ++rt) {
    Frag Af;
    const int arow = rt * 16 + lm;
    Af.q[0] = *(const uint4*)&QKVs[arow * QKVSTR + qoff + half * 16];
    Af.q[1] = *(const uint4*)&QKVs[arow * QKVSTR + qoff + half * 16 + 8];
#pragma unroll
    for (int ctk = 0; ctk < 4; ++ctk) {
      Frag Bf;
      const int brow = ctk * 16 + lm;
      const uint4* pb = (const uint4*)&QKVs[brow * QKVSTR + koff + half * 16];
      Bf.q[0] = pb[0];
      Bf.q[1] = pb[1];
      v8f z;
#pragma unroll
      for (int r = 0; r < 8; ++r) z[r] = 0.f;
      S[rt][ctk] = __builtin_amdgcn_wmma_f32_16x16x32_bf16(
          false, Af.b, false, Bf.b, (short)0, z, false, false);
    }
  }

  // precomputed rel-pos bias + shift mask (both L2-resident)
  const float* bx   = biasx + h * NSQ;
  const float* mrow = mask + (size_t)(b & 63) * NSQ;
  for (int rt = 0; rt < 4; ++rt)
#pragma unroll
    for (int ctk = 0; ctk < 4; ++ctk)
#pragma unroll
      for (int r = 0; r < 8; ++r) {
        const int n  = rt * 16 + r + half * 8;
        const int mc = ctk * 16 + lm;
        float s = S[rt][ctk][r];
        if (mc < WS_N) {
          if (n < WS_N) {
            int nm = n * WS_N + mc;
            s += bx[nm] + mrow[nm];
          }
        } else {
          s = -1.0e30f;                       // padded key columns -> p == 0
        }
        S[rt][ctk][r] = s;
      }

  // in-register softmax: fmax/add across 4 col-tiles + shfl over 16-lane half
  unsigned short* Ph = Ps + h * 64 * PSTR;
  for (int rt = 0; rt < 4; ++rt) {
#pragma unroll
    for (int r = 0; r < 8; ++r) {
      float mx = fmaxf(fmaxf(S[rt][0][r], S[rt][1][r]),
                       fmaxf(S[rt][2][r], S[rt][3][r]));
#pragma unroll
      for (int off = 8; off >= 1; off >>= 1)
        mx = fmaxf(mx, __shfl_xor(mx, off, 32));
      float sum = 0.f;
#pragma unroll
      for (int ctk = 0; ctk < 4; ++ctk) {
        float p = __expf(S[rt][ctk][r] - mx);
        S[rt][ctk][r] = p;
        sum += p;
      }
#pragma unroll
      for (int off = 8; off >= 1; off >>= 1)
        sum += __shfl_xor(sum, off, 32);
      const float rinv = 1.0f / sum;
      const int n = rt * 16 + r + half * 8;
#pragma unroll
      for (int ctk = 0; ctk < 4; ++ctk)
        Ph[n * PSTR + ctk * 16 + lm] = f2bf(S[rt][ctk][r] * rinv);
    }
  }

  // O = P @ V   ([64,64] x [64,32]); V from transposed LDS copy
  for (int rt = 0; rt < 4; ++rt) {
    const int arow = rt * 16 + lm;
#pragma unroll
    for (int ctd = 0; ctd < 2; ++ctd) {
      v8f acc;
#pragma unroll
      for (int r = 0; r < 8; ++r) acc[r] = 0.f;
      const int vd = h * HD + ctd * 16 + lm;
#pragma unroll
      for (int km = 0; km < 2; ++km) {
        const int k0 = km * 32;
        Frag Af;
        Af.q[0] = *(const uint4*)&Ph[arow * PSTR + k0 + half * 16];
        Af.q[1] = *(const uint4*)&Ph[arow * PSTR + k0 + half * 16 + 8];
        Frag Bf;
        const uint4* pv = (const uint4*)&Vt[vd * VTSTR + k0 + half * 16];
        Bf.q[0] = pv[0];
        Bf.q[1] = pv[1];
        acc = __builtin_amdgcn_wmma_f32_16x16x32_bf16(
            false, Af.b, false, Bf.b, (short)0, acc, false, false);
      }
#pragma unroll
      for (int r = 0; r < 8; ++r) {
        int n = rt * 16 + r + half * 8;
        Xs[n * XSTR + h * HD + ctd * 16 + lm] = f2bf(acc[r]);
      }
    }
  }
  __syncthreads();

  // ---- Phase 3: out = O @ proj_w^T + proj_b
  float* ob = out + (size_t)b * WS_N * CDIM;
  for (int cti = 0; cti < 2; ++cti) {
    const int ct = wv * 2 + cti;
    const int cbase = ct * 16;
    Frag Bf[8];
    const unsigned short* wrow = wp + (size_t)(cbase + lm) * CDIM + half * 16;
#pragma unroll
    for (int kk = 0; kk < 8; ++kk) {
      const uint4* p = (const uint4*)(wrow + kk * 32);
      Bf[kk].q[0] = p[0];
      Bf[kk].q[1] = p[1];
    }

    const float bias = proj_b[cbase + lm];
    for (int rt = 0; rt < 4; ++rt) {
      v8f acc;
#pragma unroll
      for (int r = 0; r < 8; ++r) acc[r] = bias;
      const int arow = rt * 16 + lm;
#pragma unroll
      for (int kk = 0; kk < 8; ++kk) {
        Frag Af;
        Af.q[0] = *(const uint4*)&Xs[arow * XSTR + kk * 32 + half * 16];
        Af.q[1] = *(const uint4*)&Xs[arow * XSTR + kk * 32 + half * 16 + 8];
        acc = __builtin_amdgcn_wmma_f32_16x16x32_bf16(
            false, Af.b, false, Bf[kk].b, (short)0, acc, false, false);
      }
#pragma unroll
      for (int r = 0; r < 8; ++r) {
        int n = rt * 16 + r + half * 8;
        if (n < WS_N) ob[(size_t)n * CDIM + cbase + lm] = acc[r];
      }
    }
  }
}

extern "C" void kernel_launch(void* const* d_in, const int* in_sizes, int n_in,
                              void* d_out, int out_size, void* d_ws, size_t ws_size,
                              hipStream_t stream) {
  const float* x      = (const float*)d_in[0];
  const float* mask   = (const float*)d_in[1];
  const float* qkv_w  = (const float*)d_in[2];
  const float* qkv_b  = (const float*)d_in[3];
  const float* proj_w = (const float*)d_in[4];
  const float* proj_b = (const float*)d_in[5];
  const float* btab   = (const float*)d_in[6];
  float* out = (float*)d_out;

  unsigned short* wq = (unsigned short*)d_ws;                 // 393,216 B
  unsigned short* wp = wq + QKVW_ELEMS;                       // 131,072 B
  float* bx = (float*)(wp + PROJW_ELEMS);                     //  76,832 B

  const int B = in_sizes[0] / (WS_N * CDIM);   // 4096 windows

  static bool attr_set = false;   // idempotent host-side attribute, not a stream op
  if (!attr_set) {
    hipFuncSetAttribute((const void*)swin_window_attn_fused,
                        hipFuncAttributeMaxDynamicSharedMemorySize, SMEM_BYTES);
    attr_set = true;
  }

  // tiny pre-passes: bf16 weights + expanded rel-pos bias (L2-resident after)
  const int cvt_threads = (QKVW_ELEMS + PROJW_ELEMS) / 4;     // 65,536
  hipLaunchKernelGGL(cvt_weights, dim3(cvt_threads / 256), dim3(256), 0, stream,
                     qkv_w, proj_w, wq, wp);
  hipLaunchKernelGGL(expand_bias, dim3((BIASX_ELEMS + 255) / 256), dim3(256), 0,
                     stream, btab, bx);

  hipLaunchKernelGGL(swin_window_attn_fused, dim3(B), dim3(256), SMEM_BYTES,
                     stream, x, mask, wq, qkv_b, wp, proj_b, bx, out);
}
